// LSTM_283467841928
// MI455X (gfx1250) — matrix operand order repaired and checked by reading, not compiled, not run on hardware
//
#include <hip/hip_runtime.h>

typedef __bf16 bf16_t;
typedef bf16_t v16bf __attribute__((ext_vector_type(16)));
typedef bf16_t v8bf  __attribute__((ext_vector_type(8)));
typedef float  v8f   __attribute__((ext_vector_type(8)));
typedef int    v4i   __attribute__((__vector_size__(16)));
typedef __attribute__((address_space(1))) v4i* gv4i_p;   // global int4*
typedef __attribute__((address_space(3))) v4i* lv4i_p;   // LDS int4*

#define LSTM_B   64
#define LSTM_T   512
#define LSTM_D   512
#define LSTM_H   1024
#define LSTM_G   4096   // 4*H

// LDS slab geometry (per block): 4 gates x 16 cols of W^T and U^T, rows padded
#define PW   (LSTM_D + 8)                 // 520 elems  -> 1040 B pitch (4-bank stagger)
#define PU   (LSTM_H + 8)                 // 1032 elems -> 2064 B pitch
#define USLAB_OFF (64 * PW)               // W slab: 64 rows
#define SLAB_ELEMS (USLAB_OFF + 64 * PU)  // 99,328 elems = 198,656 B < 320 KB

// ---- workspace layout (bytes) ----
#define XB_OFF   ((size_t)0)
#define XB_BYTES ((size_t)LSTM_B * LSTM_T * LSTM_D * 2)   // x in bf16: 32 MB
#define WT_OFF   (XB_OFF + XB_BYTES)
#define WT_BYTES ((size_t)LSTM_G * LSTM_D * 2)            // W^T [4096][512] bf16: 4 MB
#define UT_OFF   (WT_OFF + WT_BYTES)
#define UT_BYTES ((size_t)LSTM_G * LSTM_H * 2)            // U^T [4096][1024] bf16: 8 MB
#define HB_OFF   (UT_OFF + UT_BYTES)
#define HB_BYTES ((size_t)2 * LSTM_B * LSTM_H * 2)        // h ping-pong bf16: 256 KB
#define BAR_OFF  (HB_OFF + HB_BYTES)

#if __has_builtin(__builtin_amdgcn_global_load_async_to_lds_b128) && \
    __has_builtin(__builtin_amdgcn_s_wait_asynccnt)
#define HAVE_ASYNC_LDS 1
#else
#define HAVE_ASYNC_LDS 0
#endif

__device__ __forceinline__ unsigned short f2bf(float f) {
  union { float f; unsigned u; } c; c.f = f;
  unsigned u = c.u;
  unsigned r = u + 0x7FFFu + ((u >> 16) & 1u);   // round-to-nearest-even
  return (unsigned short)(r >> 16);
}

__global__ void cvt_copy_bf16(const float* __restrict__ src,
                              unsigned short* __restrict__ dst, int n) {
  int i = blockIdx.x * blockDim.x + threadIdx.x;
  if (i < n) dst[i] = f2bf(src[i]);
}

// dst[n*K + k] = bf16(src[k*N + n])  : [K][N] f32 -> [N][K] bf16
__global__ void cvt_transpose_bf16(const float* __restrict__ src,
                                   unsigned short* __restrict__ dst,
                                   int K, int N) {
  long i = (long)blockIdx.x * blockDim.x + threadIdx.x;
  long total = (long)K * N;
  if (i >= total) return;
  int k = (int)(i % K);
  int n = (int)(i / K);
  dst[i] = f2bf(src[(long)k * N + n]);
}

__global__ void init_barrier(unsigned* cnt) {
  if (threadIdx.x == 0 && blockIdx.x == 0) *cnt = 0u;
}

// ---- WMMA tile loads (ISA 7.12.2 layouts, wave32) ----
// A 16x32 bf16 from row-major [M][>=K]
__device__ __forceinline__ v16bf load_a16x32(const unsigned short* __restrict__ base,
                                             long row_stride, int lane, int k0) {
  const unsigned short* p = base + (long)(lane & 15) * row_stride + k0 + ((lane >> 4) << 3);
  v8bf lo = *reinterpret_cast<const v8bf*>(p);
  v8bf hi = *reinterpret_cast<const v8bf*>(p + 16);
  return __builtin_shufflevector(lo, hi, 0,1,2,3,4,5,6,7,8,9,10,11,12,13,14,15);
}

// B 32x16 bf16 from an [N][K]-style slab; rowbase_lane already includes
// (column row)*pitch + (lane>>4)*8.  Works for both global and LDS pointers.
__device__ __forceinline__ v16bf load_b32x16(const unsigned short* rowbase_lane, int k0) {
  const unsigned short* p = rowbase_lane + k0;
  v8bf lo = *reinterpret_cast<const v8bf*>(p);
  v8bf hi = *reinterpret_cast<const v8bf*>(p + 16);
  return __builtin_shufflevector(lo, hi, 0,1,2,3,4,5,6,7,8,9,10,11,12,13,14,15);
}

__device__ __forceinline__ float sigmoidf_fast(float x) {
  return 1.0f / (1.0f + __expf(-x));
}

// Copy one 16-byte chunk global -> LDS (async TDM-path instruction when available).
__device__ __forceinline__ void lds_fill16(unsigned short* ldst,
                                           const unsigned short* gsrc) {
#if HAVE_ASYNC_LDS
  __builtin_amdgcn_global_load_async_to_lds_b128(
      (gv4i_p)(v4i*)(void*)gsrc,
      (lv4i_p)(v4i*)(void*)ldst, 0, 0);
#else
  *reinterpret_cast<v8bf*>(ldst) = *reinterpret_cast<const v8bf*>(gsrc);
#endif
}

// Persistent fused LSTM: 64 blocks x 128 threads = 256 waves.
// Block b owns H-columns [16b,16b+16); its 4 waves own batch tiles m0 = 0/16/32/48.
// Per-block weight slab (4 gates x 16 cols of W^T and U^T) lives in LDS for all T steps.
__global__ __launch_bounds__(128) void lstm_persistent(
    const unsigned short* __restrict__ xb,   // [B][T][D] bf16
    const unsigned short* __restrict__ Wt,   // [4H][D]  bf16 (W^T)
    const unsigned short* __restrict__ Ut,   // [4H][H]  bf16 (U^T)
    const float* __restrict__ bias,          // [4H] f32
    unsigned short* __restrict__ hbuf,       // [2][B][H] bf16 ping-pong
    float* __restrict__ out,                 // [B][H] f32
    unsigned* __restrict__ bar)
{
  __shared__ __align__(16) unsigned short slab[SLAB_ELEMS];

  const int tid  = threadIdx.x;
  const int lane = tid & 31;
  const int m0   = (tid >> 5) * 16;        // batch-row tile base (wave slot)
  const int h0   = blockIdx.x * 16;        // H-column tile base (block)
  const int ln   = lane & 15;
  const int lh   = lane >> 4;

  // ---- one-time slab fill: W^T rows (64 x 512) then U^T rows (64 x 1024) ----
  // chunk = 8 bf16 = 16 bytes
  for (int c = tid; c < 64 * (LSTM_D / 8); c += 128) {        // 4096 chunks
    int r  = c >> 6;                 // slab row 0..63  (g = r>>4, n = r&15)
    int cc = c & 63;                 // chunk within row
    int grow = (r >> 4) * LSTM_H + h0 + (r & 15);
    lds_fill16(&slab[r * PW + cc * 8], Wt + (long)grow * LSTM_D + cc * 8);
  }
  for (int c = tid; c < 64 * (LSTM_H / 8); c += 128) {        // 8192 chunks
    int r  = c >> 7;
    int cc = c & 127;
    int grow = (r >> 4) * LSTM_H + h0 + (r & 15);
    lds_fill16(&slab[USLAB_OFF + r * PU + cc * 8], Ut + (long)grow * LSTM_H + cc * 8);
  }
#if HAVE_ASYNC_LDS
  __builtin_amdgcn_s_wait_asynccnt(0);
#endif
  __syncthreads();

  // Per-gate B row base pointers into the LDS slab (lane-resolved)
  const unsigned short* lwp[4];
  const unsigned short* lup[4];
  float bg[4];
#pragma unroll
  for (int g = 0; g < 4; ++g) {
    lwp[g] = &slab[(g * 16 + ln) * PW + (lh << 3)];
    lup[g] = &slab[USLAB_OFF + (g * 16 + ln) * PU + (lh << 3)];
    bg[g]  = bias[g * LSTM_H + h0 + ln];
  }

  v8f c_st, hval;
#pragma unroll
  for (int j = 0; j < 8; ++j) { c_st[j] = 0.0f; hval[j] = 0.0f; }

  const long x_rs = (long)LSTM_T * LSTM_D;   // stride between batch rows of x
  const unsigned nblocks = gridDim.x;

  for (int t = 0; t < LSTM_T; ++t) {
    v8f acc0, acc1, acc2, acc3;
#pragma unroll
    for (int j = 0; j < 8; ++j) { acc0[j] = bg[0]; acc1[j] = bg[1]; acc2[j] = bg[2]; acc3[j] = bg[3]; }

    // gates += x_t @ W   (K = 512), B from LDS
    const unsigned short* abase_x = xb + ((long)m0 * LSTM_T + t) * LSTM_D;
    for (int k0 = 0; k0 < LSTM_D; k0 += 32) {
      v16bf a = load_a16x32(abase_x, x_rs, lane, k0);
      acc0 = __builtin_amdgcn_wmma_f32_16x16x32_bf16(false, a, false, load_b32x16(lwp[0], k0), (short)0, acc0, false, false);
      acc1 = __builtin_amdgcn_wmma_f32_16x16x32_bf16(false, a, false, load_b32x16(lwp[1], k0), (short)0, acc1, false, false);
      acc2 = __builtin_amdgcn_wmma_f32_16x16x32_bf16(false, a, false, load_b32x16(lwp[2], k0), (short)0, acc2, false, false);
      acc3 = __builtin_amdgcn_wmma_f32_16x16x32_bf16(false, a, false, load_b32x16(lwp[3], k0), (short)0, acc3, false, false);
    }

    // gates += h_{t-1} @ U   (K = 1024), B from LDS; h_0 == 0 so skip at t==0
    if (t > 0) {
      const unsigned short* abase_h =
          hbuf + (long)((t + 1) & 1) * (LSTM_B * LSTM_H) + (long)m0 * LSTM_H;
      for (int k0 = 0; k0 < LSTM_H; k0 += 32) {
        v16bf a = load_a16x32(abase_h, LSTM_H, lane, k0);
        acc0 = __builtin_amdgcn_wmma_f32_16x16x32_bf16(false, a, false, load_b32x16(lup[0], k0), (short)0, acc0, false, false);
        acc1 = __builtin_amdgcn_wmma_f32_16x16x32_bf16(false, a, false, load_b32x16(lup[1], k0), (short)0, acc1, false, false);
        acc2 = __builtin_amdgcn_wmma_f32_16x16x32_bf16(false, a, false, load_b32x16(lup[2], k0), (short)0, acc2, false, false);
        acc3 = __builtin_amdgcn_wmma_f32_16x16x32_bf16(false, a, false, load_b32x16(lup[3], k0), (short)0, acc3, false, false);
      }
    }

    // elementwise cell update (C/D layout: VGPR j holds rows m0+j (lanes 0-15) / m0+j+8)
#pragma unroll
    for (int j = 0; j < 8; ++j) {
      float iv = sigmoidf_fast(acc0[j]);
      float fv = sigmoidf_fast(acc1[j]);
      float cb = tanhf(acc2[j]);
      float ov = sigmoidf_fast(acc3[j]);
      float cn = fv * c_st[j] + iv * cb;
      c_st[j] = cn;
      hval[j] = ov * tanhf(cn);
    }

    if (t < LSTM_T - 1) {
      unsigned short* hout = hbuf + (long)(t & 1) * (LSTM_B * LSTM_H);
#pragma unroll
      for (int j = 0; j < 8; ++j) {
        int row = m0 + j + (lh << 3);
        hout[(long)row * LSTM_H + h0 + ln] = f2bf(hval[j]);
      }
      __threadfence();
      // grid-wide barrier (monotonic counter; re-zeroed per launch by init_barrier)
      __syncthreads();
      if (threadIdx.x == 0) {
        atomicAdd(bar, 1u);
        const unsigned target = nblocks * (unsigned)(t + 1);
        while (__atomic_load_n(bar, __ATOMIC_RELAXED) < target) {
          __builtin_amdgcn_s_sleep(2);
        }
      }
      __syncthreads();
      __threadfence();
    }
  }

  // final h_T -> d_out (f32)
#pragma unroll
  for (int j = 0; j < 8; ++j) {
    int row = m0 + j + (lh << 3);
    out[(long)row * LSTM_H + h0 + ln] = hval[j];
  }
}

extern "C" void kernel_launch(void* const* d_in, const int* in_sizes, int n_in,
                              void* d_out, int out_size, void* d_ws, size_t ws_size,
                              hipStream_t stream) {
  const float* x = (const float*)d_in[0];   // [64][512][512]
  const float* W = (const float*)d_in[1];   // [512][4096]
  const float* U = (const float*)d_in[2];   // [1024][4096]
  const float* b = (const float*)d_in[3];   // [4096]

  char* ws = (char*)d_ws;
  unsigned short* xb  = (unsigned short*)(ws + XB_OFF);
  unsigned short* Wt  = (unsigned short*)(ws + WT_OFF);
  unsigned short* Ut  = (unsigned short*)(ws + UT_OFF);
  unsigned short* hb  = (unsigned short*)(ws + HB_OFF);
  unsigned*       bar = (unsigned*)(ws + BAR_OFF);

  const int nx = LSTM_B * LSTM_T * LSTM_D;          // 16,777,216
  const long nw = (long)LSTM_D * LSTM_G;            //  2,097,152
  const long nu = (long)LSTM_H * LSTM_G;            //  4,194,304

  cvt_copy_bf16<<<(nx + 255) / 256, 256, 0, stream>>>(x, xb, nx);
  cvt_transpose_bf16<<<(unsigned)((nw + 255) / 256), 256, 0, stream>>>(W, Wt, LSTM_D, LSTM_G);
  cvt_transpose_bf16<<<(unsigned)((nu + 255) / 256), 256, 0, stream>>>(U, Ut, LSTM_H, LSTM_G);
  init_barrier<<<1, 64, 0, stream>>>(bar);

  lstm_persistent<<<64, 128, 0, stream>>>(xb, Wt, Ut, b, hb, (float*)d_out, bar);
}
